// MultiHeadAttention_77996606095869
// MI455X (gfx1250) — matrix operand order repaired and checked by reading, not compiled, not run on hardware
//
#include <hip/hip_runtime.h>
#include <hip/hip_bf16.h>

// ---------------------------------------------------------------------------
// MultiHeadAttention for MI455X (gfx1250): bf16 WMMA everywhere, f32 accum.
// Pipeline: pad-bias -> QKV bf16 WMMA GEMMs (head-split; V pre-transposed)
//           -> flash attention (QK^T + online softmax + PV, all WMMA)
//           -> output WMMA GEMM.
// CDNA5 specifics: v_wmma_f32_16x16x32_bf16, b128 fragment loads from padded
// LDS (stride 80B, 16B aligned), v_cvt_pk_bf16_f32 packing, and
// global_load_async_to_lds_b128 (ASYNCcnt) for pure-bf16 tile staging.
// ---------------------------------------------------------------------------

typedef __attribute__((ext_vector_type(16))) __bf16 v16bf;
typedef __attribute__((ext_vector_type(8)))  float  v8f;

#define B_   2
#define T_   2048
#define D_   512
#define DM_  64
#define H_   16
#define DMH_ 1024
#define LS   40   // LDS row stride in bf16 elems: 80 B, 16B aligned, bank-safe

// ---- feature guards --------------------------------------------------------
#if defined(__has_builtin)
#if __has_builtin(__builtin_amdgcn_cvt_pk_bf16_f32)
#define USE_CVT_PK 1
#endif
#if __has_builtin(__builtin_amdgcn_global_load_async_to_lds_b128) && \
    __has_builtin(__builtin_amdgcn_s_wait_asynccnt)
#define USE_ASYNC_LDS 1
#endif
#endif

__device__ __forceinline__ unsigned short f2bf(float x) {
  union { float f; unsigned int u; } c; c.f = x;
  return (unsigned short)((c.u + 0x7FFFu + ((c.u >> 16) & 1u)) >> 16);  // RNE
}

// Pack two f32 -> two bf16 in one u32 (one v_cvt_pk_bf16_f32 when available).
__device__ __forceinline__ unsigned int pk_bf16(float a, float b) {
#ifdef USE_CVT_PK
  auto r = __builtin_amdgcn_cvt_pk_bf16_f32(a, b);
  union { decltype(r) v; unsigned int u; } c; c.v = r;
  return c.u;
#else
  return (unsigned int)f2bf(a) | ((unsigned int)f2bf(b) << 16);
#endif
}

#ifdef USE_ASYNC_LDS
// Builtin expects int4 pointers: (global int4*, lds int4*, imm off, imm cpol).
typedef int v4i_gcc __attribute__((vector_size(16)));
typedef __attribute__((address_space(1))) v4i_gcc* as1_v4i;
typedef __attribute__((address_space(3))) v4i_gcc* as3_v4i;
__device__ __forceinline__ void async_cp_b128(const unsigned short* g,
                                              unsigned short* l) {
  __builtin_amdgcn_global_load_async_to_lds_b128((as1_v4i)g, (as3_v4i)l, 0, 0);
}
#endif

__device__ __forceinline__ v8f zero8() {
  v8f z;
#pragma unroll
  for (int i = 0; i < 8; ++i) z[i] = 0.0f;
  return z;
}

union Frag { unsigned int u[8]; v16bf v; };

__device__ __forceinline__ v8f wmma_bf16(const Frag& a, const Frag& b, v8f c) {
  return __builtin_amdgcn_wmma_f32_16x16x32_bf16(
      false, a.v, false, b.v, (short)0, c, false, false);
}

// A-frag (16x32, 16-bit): lane holds row lane%16; K runs {hi*8..hi*8+7} and
// {16+hi*8..}; both are 16B-contiguous -> two b128 loads from the row base.
__device__ __forceinline__ void ld_afrag(Frag& f, const unsigned short* row, int hi) {
  uint4 a = *reinterpret_cast<const uint4*>(row + hi * 8);
  uint4 b = *reinterpret_cast<const uint4*>(row + 16 + hi * 8);
  f.u[0] = a.x; f.u[1] = a.y; f.u[2] = a.z; f.u[3] = a.w;
  f.u[4] = b.x; f.u[5] = b.y; f.u[6] = b.z; f.u[7] = b.w;
}
// B-frag (32x16, 16-bit): lane holds col lane%16; K runs {hi*8..hi*8+7} and
// {16+hi*8..} -> two b128 loads from the (transposed) row base.
__device__ __forceinline__ void ld_bfrag(Frag& f, const unsigned short* row, int hi) {
  const unsigned short* p = row + hi * 8;
  uint4 a = *reinterpret_cast<const uint4*>(p);
  uint4 b = *reinterpret_cast<const uint4*>(p + 16);
  f.u[0] = a.x; f.u[1] = a.y; f.u[2] = a.z; f.u[3] = a.w;
  f.u[4] = b.x; f.u[5] = b.y; f.u[6] = b.z; f.u[7] = b.w;
}

// ---------------------------------------------------------------------------
// Kernel 1: key padding bias. padbias[b*T+t] = (sum_d mask == 0) ? -1e9 : 0
// ---------------------------------------------------------------------------
__global__ void __launch_bounds__(128)
pad_kernel(const float* __restrict__ mask, float* __restrict__ padbias) {
  __shared__ float red[128];
  const int row = blockIdx.x;
  const float* m = mask + (size_t)row * D_;
  float s = 0.0f;
  for (int i = threadIdx.x; i < D_; i += 128) s += m[i];
  red[threadIdx.x] = s;
  __syncthreads();
#pragma unroll
  for (int off = 64; off > 0; off >>= 1) {
    if (threadIdx.x < off) red[threadIdx.x] += red[threadIdx.x + off];
    __syncthreads();
  }
  if (threadIdx.x == 0) padbias[row] = (red[0] == 0.0f) ? -1e9f : 0.0f;
}

// ---------------------------------------------------------------------------
// Kernel 2: QKV projection. X[4096,512] f32 x W[512,1024] f32 + bias, scaled,
// bf16 output. Projection column c: h = c%16, dm = c/16.
//   layout 0 (Q,K): Out[((b*H+h)*T + t)*DM + dm]
//   layout 1 (V):   Out[((b*H+h)*DM + dm)*T + t]   (pre-transposed for PV)
// ---------------------------------------------------------------------------
__global__ void __launch_bounds__(128)
gemm_qkv_kernel(const float* __restrict__ X, const float* __restrict__ W,
                const float* __restrict__ bias, unsigned short* __restrict__ Out,
                float scale, int layout) {
  __shared__ __align__(16) unsigned short Xs[64 * LS];    // [row][k]
  __shared__ __align__(16) unsigned short Wst[64 * LS];   // transposed [n][k]
  const int tid = threadIdx.x;
  const int wave = tid >> 5, lane = tid & 31;
  const int lane16 = lane & 15, hi = lane >> 4;
  const int m0 = blockIdx.x * 64;
  const int n0 = blockIdx.y * 64;

  v8f acc[4];
#pragma unroll
  for (int i = 0; i < 4; ++i) acc[i] = zero8();

  for (int kb = 0; kb < D_; kb += 32) {
    __syncthreads();
#pragma unroll
    for (int e = tid; e < 1024; e += 128) {      // X tile 64x32 -> bf16 pairs
      const int r = e >> 4, kp = (e & 15) * 2;
      const float2 x2 = *reinterpret_cast<const float2*>(
          &X[(size_t)(m0 + r) * D_ + kb + kp]);
      *reinterpret_cast<unsigned int*>(&Xs[r * LS + kp]) = pk_bf16(x2.x, x2.y);
    }
#pragma unroll
    for (int e = tid; e < 1024; e += 128) {      // W tile 32x64 -> bf16^T
      const int n = e & 63, kp = (e >> 6) * 2;
      const float w0 = W[(size_t)(kb + kp) * DMH_ + n0 + n];
      const float w1 = W[(size_t)(kb + kp + 1) * DMH_ + n0 + n];
      *reinterpret_cast<unsigned int*>(&Wst[n * LS + kp]) = pk_bf16(w0, w1);
    }
    __syncthreads();

    Frag a;
    ld_afrag(a, &Xs[(wave * 16 + lane16) * LS], hi);
#pragma unroll
    for (int nt = 0; nt < 4; ++nt) {
      Frag bf;
      ld_bfrag(bf, &Wst[(nt * 16 + lane16) * LS], hi);
      acc[nt] = wmma_bf16(a, bf, acc[nt]);
    }
  }

#pragma unroll
  for (int nt = 0; nt < 4; ++nt) {
    const int col = n0 + nt * 16 + lane16;
    const int hh = col & (H_ - 1);
    const int dm = col >> 4;
    const float bb = bias[col];
#pragma unroll
    for (int r = 0; r < 8; ++r) {
      const int row = m0 + wave * 16 + r + 8 * hi;   // C-layout: M = r + 8*hi
      const int b = row >> 11, t = row & (T_ - 1);
      const unsigned short val = f2bf((acc[nt][r] + bb) * scale);
      if (layout == 0)
        Out[(size_t)((b * H_ + hh) * T_ + t) * DM_ + dm] = val;
      else
        Out[((size_t)(b * H_ + hh) * DM_ + dm) * T_ + t] = val;
    }
  }
}

// ---------------------------------------------------------------------------
// Kernel 3: flash attention per (b,h). 4 waves, 64 query rows, 32-key tiles.
// Q frags preloaded (two b128 pairs); K frags straight from global; V tile
// async-DMA'd from the pre-transposed [dm][t] layout straight into LDS;
// P transposed C->A layout via per-wave LDS.
// Output merged layout: Obuf[(b*T + t)*1024 + h*64 + dm]  (bf16).
// ---------------------------------------------------------------------------
__global__ void __launch_bounds__(128)
attn_kernel(const unsigned short* __restrict__ Qh, const unsigned short* __restrict__ Kh,
            const unsigned short* __restrict__ Vh, const float* __restrict__ padbias,
            unsigned short* __restrict__ Obuf) {
  __shared__ __align__(16) unsigned short Vt[64 * LS];      // [dm][key]
  __shared__ __align__(16) unsigned short Plds[4][16 * LS]; // per-wave P
  const int bh = blockIdx.x;                    // b*H + h
  const int b = bh >> 4, h = bh & 15;
  const int q0 = blockIdx.y * 64;
  const int tid = threadIdx.x;
  const int wave = tid >> 5, lane = tid & 31;
  const int lane16 = lane & 15, hi = lane >> 4;

  const unsigned short* Qp = Qh + (size_t)bh * T_ * DM_;
  const unsigned short* Kp = Kh + (size_t)bh * T_ * DM_;
  const unsigned short* Vp = Vh + (size_t)bh * DM_ * T_;    // transposed [dm][t]
  const float* pbias = padbias + (size_t)b * T_;

  // Preload Q fragments: 16 rows x 64 dm = two 16x32 A-frags (Q pre-scaled).
  Frag qa[2];
  {
    const unsigned short* qrow = Qp + (size_t)(q0 + wave * 16 + lane16) * DM_;
    ld_afrag(qa[0], qrow, hi);
    ld_afrag(qa[1], qrow + 32, hi);
  }

  v8f oacc[4];
#pragma unroll
  for (int i = 0; i < 4; ++i) oacc[i] = zero8();
  float mrow[8], lrow[8];
#pragma unroll
  for (int r = 0; r < 8; ++r) { mrow[r] = -3.0e38f; lrow[r] = 0.0f; }

  for (int k0t = 0; k0t < T_; k0t += 32) {
    __syncthreads();
    // V tile: 64 dm-rows x 32 keys, 64B per row, already transposed in global.
#ifdef USE_ASYNC_LDS
#pragma unroll
    for (int e = tid; e < 256; e += 128) {        // 256 x b128 async DMA
      const int n = e >> 2, c = (e & 3) * 8;
      async_cp_b128(Vp + (size_t)n * T_ + k0t + c, &Vt[n * LS + c]);
    }
    __builtin_amdgcn_s_wait_asynccnt(0);
#else
#pragma unroll
    for (int e = tid; e < 256; e += 128) {
      const int n = e >> 2, c = (e & 3) * 8;
      *reinterpret_cast<uint4*>(&Vt[n * LS + c]) =
          *reinterpret_cast<const uint4*>(Vp + (size_t)n * T_ + k0t + c);
    }
#endif
    __syncthreads();

    // S = Q K^T : two 16-key n-tiles, two 32-dm K-steps.
    v8f sacc[2];
    sacc[0] = zero8(); sacc[1] = zero8();
#pragma unroll
    for (int ks = 0; ks < 2; ++ks) {
#pragma unroll
      for (int nt = 0; nt < 2; ++nt) {
        Frag kf;
        ld_bfrag(kf, Kp + (size_t)(k0t + nt * 16 + lane16) * DM_ + ks * 32, hi);
        sacc[nt] = wmma_bf16(qa[ks], kf, sacc[nt]);
      }
    }

    const float pb0 = pbias[k0t + lane16];
    const float pb1 = pbias[k0t + 16 + lane16];

    // Online softmax per row (row data: N = lane%16 across the 16-lane group).
#pragma unroll
    for (int r = 0; r < 8; ++r) {
      float s0 = sacc[0][r] + pb0;
      float s1 = sacc[1][r] + pb1;
      float mx = fmaxf(s0, s1);
#pragma unroll
      for (int m = 8; m >= 1; m >>= 1) mx = fmaxf(mx, __shfl_xor(mx, m, 32));
      const float mn = fmaxf(mrow[r], mx);
      const float alpha = __expf(mrow[r] - mn);
      const float p0 = __expf(s0 - mn);
      const float p1 = __expf(s1 - mn);
      float rs = p0 + p1;
#pragma unroll
      for (int m = 8; m >= 1; m >>= 1) rs += __shfl_xor(rs, m, 32);
      lrow[r] = lrow[r] * alpha + rs;
      mrow[r] = mn;
#pragma unroll
      for (int nt = 0; nt < 4; ++nt) oacc[nt][r] *= alpha;
      const int prow = r + 8 * hi;
      Plds[wave][prow * LS + lane16]      = f2bf(p0);
      Plds[wave][prow * LS + 16 + lane16] = f2bf(p1);
    }

    // Reload P as A-fragment (LDS ops from the same wave are in-order).
    Frag pa;
    ld_afrag(pa, &Plds[wave][lane16 * LS], hi);

    // O += P * V  (four dm n-tiles)
#pragma unroll
    for (int nt = 0; nt < 4; ++nt) {
      Frag vf;
      ld_bfrag(vf, &Vt[(nt * 16 + lane16) * LS], hi);
      oacc[nt] = wmma_bf16(pa, vf, oacc[nt]);
    }
  }

  // Normalize and write merged (nhead, d_model) layout.
#pragma unroll
  for (int r = 0; r < 8; ++r) {
    const float linv = 1.0f / lrow[r];
    const int t = q0 + wave * 16 + r + 8 * hi;
    const size_t rowoff = ((size_t)b * T_ + t) * DMH_ + h * DM_;
#pragma unroll
    for (int nt = 0; nt < 4; ++nt)
      Obuf[rowoff + nt * 16 + lane16] = f2bf(oacc[nt][r] * linv);
  }
}

// ---------------------------------------------------------------------------
// Kernel 4: output projection. O[4096,1024] bf16 x Wo[1024,512] f32 + bo ->
// f32 d_out. A tile is a pure bf16 copy -> async DMA to LDS.
// ---------------------------------------------------------------------------
__global__ void __launch_bounds__(128)
gemm_out_kernel(const unsigned short* __restrict__ A, const float* __restrict__ W,
                const float* __restrict__ bias, float* __restrict__ Out) {
  __shared__ __align__(16) unsigned short As[64 * LS];
  __shared__ __align__(16) unsigned short Wst[64 * LS];
  const int tid = threadIdx.x;
  const int wave = tid >> 5, lane = tid & 31;
  const int lane16 = lane & 15, hi = lane >> 4;
  const int m0 = blockIdx.x * 64;
  const int n0 = blockIdx.y * 64;

  v8f acc[4];
#pragma unroll
  for (int i = 0; i < 4; ++i) acc[i] = zero8();

  for (int kb = 0; kb < DMH_; kb += 32) {
    __syncthreads();
#ifdef USE_ASYNC_LDS
#pragma unroll
    for (int e = tid; e < 256; e += 128) {       // A tile 64x32 bf16, async
      const int r = e >> 2, c = (e & 3) * 8;
      async_cp_b128(A + (size_t)(m0 + r) * DMH_ + kb + c, &As[r * LS + c]);
    }
#else
#pragma unroll
    for (int e = tid; e < 256; e += 128) {
      const int r = e >> 2, c = (e & 3) * 8;
      *reinterpret_cast<uint4*>(&As[r * LS + c]) =
          *reinterpret_cast<const uint4*>(A + (size_t)(m0 + r) * DMH_ + kb + c);
    }
#endif
#pragma unroll
    for (int e = tid; e < 1024; e += 128) {      // Wo tile 32x64 -> bf16^T
      const int n = e & 63, kp = (e >> 6) * 2;
      const float w0 = W[(size_t)(kb + kp) * D_ + n0 + n];
      const float w1 = W[(size_t)(kb + kp + 1) * D_ + n0 + n];
      *reinterpret_cast<unsigned int*>(&Wst[n * LS + kp]) = pk_bf16(w0, w1);
    }
#ifdef USE_ASYNC_LDS
    __builtin_amdgcn_s_wait_asynccnt(0);
#endif
    __syncthreads();

    Frag a;
    ld_afrag(a, &As[(wave * 16 + lane16) * LS], hi);
#pragma unroll
    for (int nt = 0; nt < 4; ++nt) {
      Frag bf;
      ld_bfrag(bf, &Wst[(nt * 16 + lane16) * LS], hi);
      acc[nt] = wmma_bf16(a, bf, acc[nt]);
    }
  }

#pragma unroll
  for (int nt = 0; nt < 4; ++nt) {
    const int col = n0 + nt * 16 + lane16;
    const float bb = bias[col];
#pragma unroll
    for (int r = 0; r < 8; ++r) {
      const int row = m0 + wave * 16 + r + 8 * hi;
      Out[(size_t)row * D_ + col] = acc[nt][r] + bb;
    }
  }
}

// ---------------------------------------------------------------------------
extern "C" void kernel_launch(void* const* d_in, const int* in_sizes, int n_in,
                              void* d_out, int out_size, void* d_ws, size_t ws_size,
                              hipStream_t stream) {
  const float* data = (const float*)d_in[0];
  const float* mask = (const float*)d_in[1];
  const float* wq   = (const float*)d_in[2];
  const float* bq   = (const float*)d_in[3];
  const float* wk   = (const float*)d_in[4];
  const float* bk   = (const float*)d_in[5];
  const float* wv   = (const float*)d_in[6];
  const float* bv   = (const float*)d_in[7];
  const float* wo   = (const float*)d_in[8];
  const float* bo   = (const float*)d_in[9];
  float* out = (float*)d_out;

  // Workspace layout (bf16 regions 8 MB each, padbias 16 KB; total ~32 MB).
  const size_t nqkv = (size_t)B_ * H_ * T_ * DM_;      // 4,194,304 elems
  unsigned short* Qh = (unsigned short*)d_ws;
  unsigned short* Kh = Qh + nqkv;
  unsigned short* Vh = Kh + nqkv;                       // transposed [bh][dm][t]
  unsigned short* Ob = Vh + nqkv;
  float* pbias = (float*)(Ob + (size_t)B_ * T_ * DMH_);

  pad_kernel<<<B_ * T_, 128, 0, stream>>>(mask, pbias);
  gemm_qkv_kernel<<<dim3(64, 16), 128, 0, stream>>>(data, wq, bq, Qh, 0.125f, 0);
  gemm_qkv_kernel<<<dim3(64, 16), 128, 0, stream>>>(data, wk, bk, Kh, 1.0f, 0);
  gemm_qkv_kernel<<<dim3(64, 16), 128, 0, stream>>>(data, wv, bv, Vh, 1.0f, 1);
  attn_kernel<<<dim3(B_ * H_, T_ / 64), 128, 0, stream>>>(Qh, Kh, Vh, pbias, Ob);
  gemm_out_kernel<<<dim3(64, 8), 128, 0, stream>>>(Ob, wo, bo, out);
}
